// LATTE_78932908966055
// MI455X (gfx1250) — compile-verified, matrix operand order checked by reading, not run on hardware
//
#include <hip/hip_runtime.h>

// ---------------------------------------------------------------------------
// LATTE forward, MI455X (gfx1250, wave32).
//
// Algebraic collapse (exact up to fp rounding of softmax-weight sums):
//   message = v[dst]  =>  h_m[n] = v[n] * [indeg_m(n) > 0]
// so the entire edge softmax reduces to per-(relation,node) in-degree flags,
// and k/attn_l/attn_r/Wl/bl are dead. Remaining hot path:
//   v = feat @ Wr + br        (bf16 WMMA, 6.6 GFLOP)
//   beta attention over heads (axis=2 softmax, per reference) + LN + ReLU
// Roofline: ~215 MB streamed => ~10us floor at 23.3 TB/s; GEMM rides the
// v_wmma_f32_16x16x32_bf16 path with double-buffered async global->LDS
// staging (ASYNCcnt ping-pong: wait <=4 keeps one stage in flight).
// ---------------------------------------------------------------------------

#define N_NODES  50000
#define IN_DIMK  256
#define DMODEL   256
#define NHEADS   4
#define CHEAD    64
#define NREL     3
#define NEDGE    800000
#define LN_EPS   1e-5f
#define NEG_SLOPE 0.2f

typedef __attribute__((ext_vector_type(16))) __bf16 bf16x16;
typedef __attribute__((ext_vector_type(8)))  __bf16 bf16x8;
typedef __attribute__((ext_vector_type(4)))  __bf16 bf16x4;
typedef __attribute__((ext_vector_type(8)))  float  f32x8;
typedef __attribute__((ext_vector_type(4)))  float  f32x4;
typedef __attribute__((ext_vector_type(4)))  unsigned int u32x4;
typedef __attribute__((ext_vector_type(4)))  int v4i;
typedef __attribute__((address_space(3))) v4i as3_v4i;

#if defined(__AMDGCN__) && __has_builtin(__builtin_amdgcn_global_load_async_to_lds_b128) && __has_builtin(__builtin_amdgcn_s_wait_asynccnt)
#define USE_ASYNC_LDS 1
#endif

__device__ __forceinline__ void copy16_g2l(const __bf16* g, __bf16* l) {
#ifdef USE_ASYNC_LDS
  __builtin_amdgcn_global_load_async_to_lds_b128((v4i*)(void*)g, (as3_v4i*)l, 0, 0);
#else
  *(u32x4*)l = *(const u32x4*)g;
#endif
}

// Wait until at most `n` async global->LDS ops outstanding (immediates only).
__device__ __forceinline__ void wait_async_le4() {
#ifdef USE_ASYNC_LDS
  __builtin_amdgcn_s_wait_asynccnt(4);
#endif
}
__device__ __forceinline__ void wait_async_le0() {
#ifdef USE_ASYNC_LDS
  __builtin_amdgcn_s_wait_asynccnt(0);
#endif
}

__device__ __forceinline__ float lrelu(float x) { return x > 0.f ? x : NEG_SLOPE * x; }

// ---------------------------------------------------------------------------
// Kernel 0: feat f32 -> bf16 (vectorized x4)
// ---------------------------------------------------------------------------
__global__ void cvt_feat_kernel(const float* __restrict__ f, __bf16* __restrict__ o, long n4) {
  long i = (long)blockIdx.x * blockDim.x + threadIdx.x;
  if (i >= n4) return;
  f32x4 v = ((const f32x4*)f)[i];
  ((bf16x4*)o)[i] = __builtin_convertvector(v, bf16x4);
}

// ---------------------------------------------------------------------------
// Kernel 1: WT[n][k] = (bf16) Wr[k][n]   (B fragments want column-major Wr)
// ---------------------------------------------------------------------------
__global__ void cvt_w_kernel(const float* __restrict__ w, __bf16* __restrict__ wt) {
  int t = blockIdx.x * blockDim.x + threadIdx.x;   // t = n*256 + k
  int n = t >> 8, k = t & 255;
  wt[t] = (__bf16)w[k * DMODEL + n];
}

// ---------------------------------------------------------------------------
// Kernels 2/3: in-degree flags per (relation, node)
// ---------------------------------------------------------------------------
__global__ void flags_init_kernel(float* __restrict__ flags) {
  int i = blockIdx.x * blockDim.x + threadIdx.x;
  if (i < NREL * N_NODES) flags[i] = 0.f;
}

__global__ void flags_mark_kernel(const int* __restrict__ d0, const int* __restrict__ d1,
                                  const int* __restrict__ d2, float* __restrict__ flags) {
  int e = blockIdx.x * blockDim.x + threadIdx.x;
  if (e >= NEDGE) return;
  int m = blockIdx.y;
  const int* d = (m == 0) ? d0 : ((m == 1) ? d1 : d2);
  flags[m * N_NODES + d[e]] = 1.0f;   // racy identical stores: fine
}

// ---------------------------------------------------------------------------
// Kernel 4: v = feat @ Wr + br  via v_wmma_f32_16x16x32_bf16
// Block tile 128x128, 8 waves, each wave 32x64 (2x4 WMMA accums).
// K staged through LDS in 64-wide slices; DOUBLE-BUFFERED async copies:
// stage s+1 streams into buf[(s+1)&1] while WMMAs consume buf[s&1].
// Each stage = 4 async-load instrs per wave => s_wait_asynccnt 4 retires
// exactly the older stage (async loads complete in order).
// ---------------------------------------------------------------------------
#define BM 128
#define BN 128
#define KSTEP 64
#define NSTAGE (IN_DIMK / KSTEP)   // 4
#define AST 72   // LDS row stride (bf16 elems): 64 + 8 pad, keeps 16B align + bank spread

__global__ __launch_bounds__(256) void gemm_v_wmma_kernel(
    const __bf16* __restrict__ featb,   // [N][256] bf16 row-major
    const __bf16* __restrict__ wtb,     // [256 out-cols][256 k] bf16 (Wr^T)
    const float*  __restrict__ bias,    // [256]
    float* __restrict__ vout)           // [N][256] f32
{
  __shared__ __align__(16) __bf16 Ash[2][BM * AST];
  __shared__ __align__(16) __bf16 Bsh[2][BN * AST];

  const int tid  = threadIdx.x;
  const int gr0  = blockIdx.y * BM;
  const int gc0  = blockIdx.x * BN;
  const int lane = tid & 31;
  const int wv   = tid >> 5;
  const int wr   = (wv & 3) * 32;   // wave row base in block tile
  const int wc   = (wv >> 2) * 64;  // wave col base in block tile
  const int l16  = lane & 15;
  const int half = lane >> 4;

  // per-thread staging coordinates: 512 x 16B chunks per matrix per stage,
  // 256 threads -> 2 chunks each per matrix (4 async instrs per wave/stage)
  const int row0 = tid >> 2,          c0 = tid & 3;
  const int row1 = (tid + 256) >> 2,  c1 = (tid + 256) & 3;
  int ga0 = gr0 + row0; if (ga0 >= N_NODES) ga0 = N_NODES - 1;
  int ga1 = gr0 + row1; if (ga1 >= N_NODES) ga1 = N_NODES - 1;

  auto stage = [&](int ks, int pb) {
    copy16_g2l(featb + (size_t)ga0 * IN_DIMK + ks + c0 * 8,          &Ash[pb][row0 * AST + c0 * 8]);
    copy16_g2l(wtb   + (size_t)(gc0 + row0) * IN_DIMK + ks + c0 * 8, &Bsh[pb][row0 * AST + c0 * 8]);
    copy16_g2l(featb + (size_t)ga1 * IN_DIMK + ks + c1 * 8,          &Ash[pb][row1 * AST + c1 * 8]);
    copy16_g2l(wtb   + (size_t)(gc0 + row1) * IN_DIMK + ks + c1 * 8, &Bsh[pb][row1 * AST + c1 * 8]);
  };

  f32x8 acc[2][4];
#pragma unroll
  for (int i = 0; i < 2; ++i)
#pragma unroll
    for (int j = 0; j < 4; ++j) acc[i][j] = (f32x8)0.f;

  stage(0, 0);   // prologue: stage 0 in flight

#pragma unroll
  for (int s = 0; s < NSTAGE; ++s) {
    if (s + 1 < NSTAGE) {
      stage((s + 1) * KSTEP, (s + 1) & 1);  // overlap: next stage streams in
      wait_async_le4();                     // oldest 4 (stage s) have landed
    } else {
      wait_async_le0();
    }
    __syncthreads();

    const __bf16* Ab = &Ash[s & 1][0];
    const __bf16* Bb = &Bsh[s & 1][0];
#pragma unroll
    for (int kk = 0; kk < KSTEP; kk += 32) {
      // A fragment (16x32 bf16): lane<16 holds row, K {0..7,16..23}; lane>=16: {8..15,24..31}
      bf16x16 af[2], bfr[4];
#pragma unroll
      for (int i = 0; i < 2; ++i) {
        const __bf16* p = Ab + (wr + i * 16 + l16) * AST + kk + half * 8;
        bf16x8 lo = *(const bf16x8*)p;
        bf16x8 hi = *(const bf16x8*)(p + 16);
        af[i] = __builtin_shufflevector(lo, hi, 0,1,2,3,4,5,6,7,8,9,10,11,12,13,14,15);
      }
#pragma unroll
      for (int j = 0; j < 4; ++j) {
        const __bf16* p = Bb + (wc + j * 16 + l16) * AST + kk + half * 8;
        bf16x8 lo = *(const bf16x8*)p;
        bf16x8 hi = *(const bf16x8*)(p + 16);
        bfr[j] = __builtin_shufflevector(lo, hi, 0,1,2,3,4,5,6,7,8,9,10,11,12,13,14,15);
      }
#pragma unroll
      for (int i = 0; i < 2; ++i)
#pragma unroll
        for (int j = 0; j < 4; ++j)
          acc[i][j] = __builtin_amdgcn_wmma_f32_16x16x32_bf16(
              false, af[i], false, bfr[j], (short)0, acc[i][j], false, false);
    }
    __syncthreads();   // buffer s&1 free for stage s+2 issue
  }

  // Epilogue: C/D layout -> VGPR r: lanes0-15 M=r, lanes16-31 M=r+8; N=lane&15
#pragma unroll
  for (int i = 0; i < 2; ++i) {
#pragma unroll
    for (int j = 0; j < 4; ++j) {
      int col = gc0 + wc + j * 16 + l16;
      float b = bias[col];
#pragma unroll
      for (int r = 0; r < 8; ++r) {
        int row = gr0 + wr + i * 16 + half * 8 + r;
        if (row < N_NODES) vout[(size_t)row * DMODEL + col] = acc[i][j][r] + b;
      }
    }
  }
}

// ---------------------------------------------------------------------------
// Kernel 5: per-node beta attention (softmax over HEADS, per reference axis=2)
// + LayerNorm + ReLU. One wave32 per node; lane l holds dims [8l, 8l+8).
// ---------------------------------------------------------------------------
__global__ __launch_bounds__(256) void fuse_kernel(
    const float* __restrict__ v, const float* __restrict__ flags,
    const float* __restrict__ ral, const float* __restrict__ rar,
    const float* __restrict__ gamma, const float* __restrict__ lbeta,
    float* __restrict__ out)
{
  int node = (blockIdx.x * blockDim.x + threadIdx.x) >> 5;
  int lane = threadIdx.x & 31;
  if (node >= N_NODES) return;

  const float* vp = v + (size_t)node * DMODEL + lane * 8;
  f32x4 x0 = *(const f32x4*)vp;
  f32x4 x1 = *(const f32x4*)(vp + 4);
  float x[8] = {x0.x, x0.y, x0.z, x0.w, x1.x, x1.y, x1.z, x1.w};

  int h    = lane >> 3;             // head (8 lanes per 64-wide head)
  int coff = (lane & 7) * 8;
  const float* pl = ral + h * CHEAD + coff;
  const float* pr = rar + h * CHEAD + coff;
  float dl = 0.f, dr = 0.f;
#pragma unroll
  for (int t = 0; t < 8; ++t) { dl += x[t] * pl[t]; dr += x[t] * pr[t]; }
  // reduce within 8-lane head group
#pragma unroll
  for (int m = 1; m < 8; m <<= 1) {
    dl += __shfl_xor(dl, m, 32);
    dr += __shfl_xor(dr, m, 32);
  }
  // broadcast all 4 head dots to every lane
  float dlh[4], drh[4];
#pragma unroll
  for (int j = 0; j < 4; ++j) {
    dlh[j] = __shfl(dl, j * 8, 32);
    drh[j] = __shfl(dr, j * 8, 32);
  }

  float fr[4] = { flags[node], flags[N_NODES + node], flags[2 * N_NODES + node], 1.f };

  // beta[r,h] = softmax over h of leaky_relu(alpha_l[h] + f_r * dr[h])
  // s[h] = sum_{r<3} f_r * beta[r,h] + beta[3,h]
  float s = 0.f;
#pragma unroll
  for (int r = 0; r < 4; ++r) {
    float lg[4], mx = -1e30f;
#pragma unroll
    for (int hh = 0; hh < 4; ++hh) { lg[hh] = lrelu(dlh[hh] + fr[r] * drh[hh]); mx = fmaxf(mx, lg[hh]); }
    float den = 0.f, eh = 0.f;
#pragma unroll
    for (int hh = 0; hh < 4; ++hh) { float e = __expf(lg[hh] - mx); den += e; if (hh == h) eh = e; }
    s += fr[r] * (eh / den);
  }

  // o = x * s, then LayerNorm over 256 dims + ReLU
  float o[8], lsum = 0.f;
#pragma unroll
  for (int t = 0; t < 8; ++t) { o[t] = x[t] * s; lsum += o[t]; }
#pragma unroll
  for (int m = 1; m < 32; m <<= 1) lsum += __shfl_xor(lsum, m, 32);
  float mu = lsum * (1.f / DMODEL);
  float lvar = 0.f;
#pragma unroll
  for (int t = 0; t < 8; ++t) { float d = o[t] - mu; lvar += d * d; }
#pragma unroll
  for (int m = 1; m < 32; m <<= 1) lvar += __shfl_xor(lvar, m, 32);
  float rstd = rsqrtf(lvar * (1.f / DMODEL) + LN_EPS);

  int dbase = lane * 8;
  f32x4 g0 = *(const f32x4*)(gamma + dbase), g1 = *(const f32x4*)(gamma + dbase + 4);
  f32x4 b0 = *(const f32x4*)(lbeta + dbase), b1 = *(const f32x4*)(lbeta + dbase + 4);
  float G[8] = {g0.x, g0.y, g0.z, g0.w, g1.x, g1.y, g1.z, g1.w};
  float B[8] = {b0.x, b0.y, b0.z, b0.w, b1.x, b1.y, b1.z, b1.w};
  f32x4 r0, r1;
#pragma unroll
  for (int t = 0; t < 8; ++t) {
    float y = fmaxf((o[t] - mu) * rstd * G[t] + B[t], 0.f);
    if (t < 4) r0[t] = y; else r1[t - 4] = y;
  }
  float* op = out + (size_t)node * DMODEL + dbase;
  *(f32x4*)op = r0;
  *(f32x4*)(op + 4) = r1;
}

// ---------------------------------------------------------------------------
extern "C" void kernel_launch(void* const* d_in, const int* in_sizes, int n_in,
                              void* d_out, int out_size, void* d_ws, size_t ws_size,
                              hipStream_t stream) {
  const float* feat  = (const float*)d_in[0];
  const float* Wr    = (const float*)d_in[3];
  const float* br    = (const float*)d_in[4];
  const float* ral   = (const float*)d_in[7];
  const float* rar   = (const float*)d_in[8];
  const float* gamma = (const float*)d_in[9];
  const float* lbeta = (const float*)d_in[10];
  const int*   dst0  = (const int*)d_in[12];
  const int*   dst1  = (const int*)d_in[14];
  const int*   dst2  = (const int*)d_in[16];
  float* out = (float*)d_out;

  char* ws = (char*)d_ws;
  size_t off = 0;
  auto alloc = [&](size_t bytes) -> void* {
    void* p = ws + off;
    off += (bytes + 255) & ~(size_t)255;
    return p;
  };
  __bf16* featb = (__bf16*)alloc((size_t)N_NODES * IN_DIMK * sizeof(__bf16));
  __bf16* wtb   = (__bf16*)alloc((size_t)DMODEL * IN_DIMK * sizeof(__bf16));
  float*  vbuf  = (float*) alloc((size_t)N_NODES * DMODEL * sizeof(float));
  float*  flags = (float*) alloc((size_t)NREL * N_NODES * sizeof(float));

  {
    long n4 = (long)N_NODES * IN_DIMK / 4;
    cvt_feat_kernel<<<(n4 + 255) / 256, 256, 0, stream>>>(feat, featb, n4);
  }
  cvt_w_kernel<<<(DMODEL * IN_DIMK) / 256, 256, 0, stream>>>(Wr, wtb);
  flags_init_kernel<<<(NREL * N_NODES + 255) / 256, 256, 0, stream>>>(flags);
  flags_mark_kernel<<<dim3((NEDGE + 255) / 256, NREL), 256, 0, stream>>>(dst0, dst1, dst2, flags);
  gemm_v_wmma_kernel<<<dim3(DMODEL / BN, (N_NODES + BM - 1) / BM), 256, 0, stream>>>(featb, wtb, br, vbuf);
  fuse_kernel<<<N_NODES / 8, 256, 0, stream>>>(vbuf, flags, ral, rar, gamma, lbeta, out);
}
